// _NonLocalBlockND_76407468196200
// MI455X (gfx1250) — matrix operand order repaired and checked by reading, not compiled, not run on hardware
//
#include <hip/hip_runtime.h>

// ---------------------------------------------------------------------------
// NonLocalBlockND on MI455X (gfx1250): all GEMMs on v_wmma_f32_16x16x32_bf16,
// global->LDS staging via GLOBAL_LOAD_ASYNC_TO_LDS_B128 (ASYNCcnt-tracked DMA).
// Problem: B=8, C=256, Ci=128, H=W=64 -> HW=4096, Ns=1024.
// ---------------------------------------------------------------------------

typedef __bf16 bf16_t;
typedef __attribute__((ext_vector_type(16))) __bf16 v16bf;
typedef __attribute__((ext_vector_type(8)))  __bf16 v8bf;
typedef __attribute__((ext_vector_type(8)))  float  v8f;

#define HWN 4096
#define CCH 256
#define CIH 128
#define NBATCH 8
#define NSS 1024
#define LSTR 40   // LDS row stride (32 + 8 pad) in bf16 elems -> conflict-free

__device__ __forceinline__ v16bf cat8(v8bf lo, v8bf hi) {
  return __builtin_shufflevector(lo, hi, 0,1,2,3,4,5,6,7,8,9,10,11,12,13,14,15);
}

// Async DMA: 16 bytes per lane, global -> LDS, tracked by ASYNCcnt.
__device__ __forceinline__ void async_ld16(unsigned lds_byte_addr, const void* gaddr) {
  asm volatile("global_load_async_to_lds_b128 %0, %1, off"
               :: "v"(lds_byte_addr), "v"(gaddr) : "memory");
}
__device__ __forceinline__ void wait_async0() {
  asm volatile("s_wait_asynccnt 0x0" ::: "memory");
}

// Generic batched GEMM: C[M][N] = alpha * A[M][K] @ B[K][N] (+ bias[M])
// A row-major [M][K]; B supplied K-contiguous as Bt[N][K] (column-major B).
// Block: 256 thr = 8 waves; block tile 128x128; wave tile 32x64 (2x4 WMMA).
// Requires: M%128==0, N%128==0, K%64==0 (k-loop unrolled by 2, static buffers).
template<bool OUT_F32, bool HAS_BIAS>
__global__ __launch_bounds__(256)
void wmma_gemm_bf16(const bf16_t* __restrict__ A, long sAb,
                    const bf16_t* __restrict__ Bt, long sBb,
                    void* __restrict__ Cout, long sCb,
                    const float* __restrict__ bias,
                    int M, int N, int K, float alpha)
{
  __shared__ bf16_t lA[2][128 * LSTR];
  __shared__ bf16_t lB[2][128 * LSTR];

  const int tid = threadIdx.x;
  const int b   = blockIdx.z;
  const int m0  = blockIdx.y * 128;
  const int n0  = blockIdx.x * 128;
  A  += (long)b * sAb;
  Bt += (long)b * sBb;

  const int wave = tid >> 5, lane = tid & 31;
  const int wm = wave >> 1;         // 0..3 -> M offset 32*wm
  const int wn = wave & 1;          // 0..1 -> N offset 64*wn
  const int r  = lane & 15;         // A row / B column within 16-tile
  const int h  = lane >> 4;         // half select per ISA operand layout

  v8f zero = {};
  v8f acc[2][4];
  #pragma unroll
  for (int i = 0; i < 2; ++i)
    #pragma unroll
    for (int j = 0; j < 4; ++j) acc[i][j] = zero;

  // each thread DMAs 2x16B for A and 2x16B for B per k-step
  const int rowL = tid >> 2;            // 0..63
  const int rowH = rowL + 64;           // 64..127
  const int col8 = (tid & 3) * 8;       // 0,8,16,24
  const bf16_t* gA0 = A  + (long)(m0 + rowL) * K + col8;
  const bf16_t* gA1 = A  + (long)(m0 + rowH) * K + col8;
  const bf16_t* gB0 = Bt + (long)(n0 + rowL) * K + col8;
  const bf16_t* gB1 = Bt + (long)(n0 + rowH) * K + col8;
  const int ldsL = rowL * LSTR + col8;
  const int ldsH = rowH * LSTR + col8;
  // LDS byte addresses (low 32 bits of generic shared pointer = LDS offset)
  const unsigned lAL[2] = { (unsigned)(size_t)&lA[0][ldsL], (unsigned)(size_t)&lA[1][ldsL] };
  const unsigned lAH[2] = { (unsigned)(size_t)&lA[0][ldsH], (unsigned)(size_t)&lA[1][ldsH] };
  const unsigned lBL[2] = { (unsigned)(size_t)&lB[0][ldsL], (unsigned)(size_t)&lB[1][ldsL] };
  const unsigned lBH[2] = { (unsigned)(size_t)&lB[0][ldsH], (unsigned)(size_t)&lB[1][ldsH] };

  auto issue = [&](int buf, int kElems) {
    async_ld16(lAL[buf], gA0 + kElems);
    async_ld16(lAH[buf], gA1 + kElems);
    async_ld16(lBL[buf], gB0 + kElems);
    async_ld16(lBH[buf], gB1 + kElems);
  };

  auto compute = [&](int buf) {
    // A fragment (ISA 16-bit A 16x32): lane(r,h) holds K=8h..8h+7, 16+8h..16+8h+7 of row r
    v16bf af[2];
    #pragma unroll
    for (int tm = 0; tm < 2; ++tm) {
      const bf16_t* p = &lA[buf][(32 * wm + 16 * tm + r) * LSTR];
      af[tm] = cat8(*(const v8bf*)(p + 8 * h), *(const v8bf*)(p + 16 + 8 * h));
    }
    // B fragment (ISA 16-bit B 32x16): lane(n=r,h) holds K=8h..8h+7, 16+8h..16+8h+7 of column n
    v16bf bfr[4];
    #pragma unroll
    for (int tn = 0; tn < 4; ++tn) {
      const bf16_t* p = &lB[buf][(64 * wn + 16 * tn + r) * LSTR + 8 * h];
      bfr[tn] = cat8(*(const v8bf*)p, *(const v8bf*)(p + 16));
    }
    #pragma unroll
    for (int tm = 0; tm < 2; ++tm)
      #pragma unroll
      for (int tn = 0; tn < 4; ++tn)
        acc[tm][tn] = __builtin_amdgcn_wmma_f32_16x16x32_bf16(
            false, af[tm], false, bfr[tn], (short)0, acc[tm][tn], false, false);
  };

  const int nk = K >> 5;                // even by contract
  issue(0, 0);
  wait_async0();
  __syncthreads();

  for (int kt = 0; kt < nk; kt += 2) {
    issue(1, (kt + 1) << 5);            // DMA tile kt+1 while computing kt
    compute(0);
    wait_async0();
    __syncthreads();

    if (kt + 2 < nk) issue(0, (kt + 2) << 5);
    compute(1);
    wait_async0();
    __syncthreads();
  }

  // epilogue: C/D f32 16x16 layout -> element v of lane(r,h) is (row 8h+v, col r)
  float bvals[2][8];
  if constexpr (HAS_BIAS) {
    #pragma unroll
    for (int tm = 0; tm < 2; ++tm) {
      const int mb = m0 + 32 * wm + 16 * tm + 8 * h;
      #pragma unroll
      for (int v = 0; v < 8; ++v) bvals[tm][v] = bias[mb + v];
    }
  }
  #pragma unroll
  for (int tm = 0; tm < 2; ++tm) {
    #pragma unroll
    for (int tn = 0; tn < 4; ++tn) {
      const int mb = m0 + 32 * wm + 16 * tm + 8 * h;
      const int nc = n0 + 64 * wn + 16 * tn + r;
      #pragma unroll
      for (int v = 0; v < 8; ++v) {
        float val = alpha * acc[tm][tn][v];
        if constexpr (HAS_BIAS) val += bvals[tm][v];
        const int mr = mb + v;
        if constexpr (OUT_F32)
          ((float*)Cout)[(long)b * sCb + (long)mr * N + nc] = val;
        else
          ((bf16_t*)Cout)[(long)b * sCb + (long)mr * N + nc] = (bf16_t)val;
      }
    }
  }
}

// --- weight prep: stack theta/g/phi into Wproj[384][256] bf16, biases fp32 --
__global__ __launch_bounds__(256)
void k_prep_weights(const float* __restrict__ g_w,     const float* __restrict__ g_b,
                    const float* __restrict__ theta_w, const float* __restrict__ theta_b,
                    const float* __restrict__ phi_w,   const float* __restrict__ phi_b,
                    const float* __restrict__ w_w,
                    bf16_t* __restrict__ Wproj, float* __restrict__ bproj,
                    bf16_t* __restrict__ Ww)
{
  const int i = blockIdx.x * 256 + threadIdx.x;
  if      (i < 32768)         Wproj[i] = (bf16_t)theta_w[i];           // rows [0,128)
  else if (i < 65536)         Wproj[i] = (bf16_t)g_w[i - 32768];       // rows [128,256)
  else if (i < 98304)         Wproj[i] = (bf16_t)phi_w[i - 65536];     // rows [256,384)
  else if (i < 131072)        Ww[i - 98304] = (bf16_t)w_w[i - 98304];
  else if (i < 131072 + 128)  bproj[i - 131072] = theta_b[i - 131072];
  else if (i < 131072 + 256)  bproj[i - 131072] = g_b[i - 131072 - 128];
  else if (i < 131072 + 384)  bproj[i - 131072] = phi_b[i - 131072 - 256];
}

// --- xT[b][hw][c] = bf16(x[b][c][hw]) via 32x32 LDS tile transpose ---------
__global__ __launch_bounds__(256)
void k_transpose_x(const float* __restrict__ x, bf16_t* __restrict__ xT)
{
  __shared__ float tile[32][33];
  const int b = blockIdx.z, hw0 = blockIdx.x * 32, c0 = blockIdx.y * 32;
  const int tx = threadIdx.x & 31, ty0 = threadIdx.x >> 5;
  #pragma unroll
  for (int i = 0; i < 4; ++i) {
    const int ty = ty0 + i * 8;
    tile[ty][tx] = x[((long)b * CCH + c0 + ty) * HWN + hw0 + tx];
  }
  __syncthreads();
  #pragma unroll
  for (int i = 0; i < 4; ++i) {
    const int ty = ty0 + i * 8;
    xT[((long)b * HWN + hw0 + ty) * CCH + c0 + tx] = (bf16_t)tile[tx][ty];
  }
}

// --- thetaT[b][hw][ci] = proj[b][ci][hw] (ci in [0,128)) -------------------
__global__ __launch_bounds__(256)
void k_theta_T(const bf16_t* __restrict__ proj, bf16_t* __restrict__ thetaT)
{
  __shared__ bf16_t tile[32][33];
  const int b = blockIdx.z, hw0 = blockIdx.x * 32, c0 = blockIdx.y * 32;
  const bf16_t* src = proj + (long)b * 384 * HWN;
  const int tx = threadIdx.x & 31, ty0 = threadIdx.x >> 5;
  #pragma unroll
  for (int i = 0; i < 4; ++i) {
    const int ty = ty0 + i * 8;
    tile[ty][tx] = src[(long)(c0 + ty) * HWN + hw0 + tx];
  }
  __syncthreads();
  bf16_t* dst = thetaT + (long)b * HWN * CIH;
  #pragma unroll
  for (int i = 0; i < 4; ++i) {
    const int ty = ty0 + i * 8;
    dst[(long)(hw0 + ty) * CIH + c0 + tx] = tile[tx][ty];
  }
}

// --- g_p[b][ci][m]: 2x2 maxpool of proj rows [128,256) ---------------------
__global__ __launch_bounds__(256)
void k_pool_g(const bf16_t* __restrict__ proj, bf16_t* __restrict__ g_p)
{
  const long i = (long)blockIdx.x * 256 + threadIdx.x;   // B*128*1024
  const int m = (int)(i & 1023), ci = (int)((i >> 10) & 127), b = (int)(i >> 17);
  const int mx = m & 31, my = m >> 5;
  const bf16_t* p = proj + ((long)b * 384 + 128 + ci) * HWN + (2 * my) * 64 + 2 * mx;
  const float v = fmaxf(fmaxf((float)p[0], (float)p[1]),
                        fmaxf((float)p[64], (float)p[65]));
  g_p[i] = (bf16_t)v;
}

// --- phiT[b][m][ci]: 2x2 maxpool of proj rows [256,384), transposed --------
__global__ __launch_bounds__(256)
void k_pool_phiT(const bf16_t* __restrict__ proj, bf16_t* __restrict__ phiT)
{
  const long i = (long)blockIdx.x * 256 + threadIdx.x;   // B*1024*128
  const int ci = (int)(i & 127), m = (int)((i >> 7) & 1023), b = (int)(i >> 17);
  const int mx = m & 31, my = m >> 5;
  const bf16_t* p = proj + ((long)b * 384 + 256 + ci) * HWN + (2 * my) * 64 + 2 * mx;
  const float v = fmaxf(fmaxf((float)p[0], (float)p[1]),
                        fmaxf((float)p[64], (float)p[65]));
  phiT[i] = (bf16_t)v;
}

// --- BatchNorm stats: mean / rsqrt(var+eps) per channel over B*H*W ---------
__global__ __launch_bounds__(256)
void k_bn_stats(const float* __restrict__ z, float* __restrict__ meanp,
                float* __restrict__ invp)
{
  const int c = blockIdx.x;
  float s = 0.f, ss = 0.f;
  for (int i = threadIdx.x; i < NBATCH * HWN; i += 256) {
    const int b = i >> 12, hw = i & (HWN - 1);
    const float v = z[((long)b * CCH + c) * HWN + hw];
    s += v; ss += v * v;
  }
  __shared__ float sh1[256], sh2[256];
  sh1[threadIdx.x] = s; sh2[threadIdx.x] = ss;
  __syncthreads();
  for (int st = 128; st > 0; st >>= 1) {
    if (threadIdx.x < st) {
      sh1[threadIdx.x] += sh1[threadIdx.x + st];
      sh2[threadIdx.x] += sh2[threadIdx.x + st];
    }
    __syncthreads();
  }
  if (threadIdx.x == 0) {
    const float n = (float)(NBATCH * HWN);
    const float m = sh1[0] / n;
    const float var = sh2[0] / n - m * m;
    meanp[c] = m;
    invp[c]  = rsqrtf(var + 1e-5f);
  }
}

// --- out = gamma*(z-mean)*inv + beta + x -----------------------------------
__global__ __launch_bounds__(256)
void k_bn_apply(const float* __restrict__ z, const float* __restrict__ x,
                const float* __restrict__ gamma, const float* __restrict__ beta,
                const float* __restrict__ meanp, const float* __restrict__ invp,
                float* __restrict__ out)
{
  const long i = (long)blockIdx.x * 256 + threadIdx.x;
  const int c = (int)((i >> 12) & (CCH - 1));
  out[i] = gamma[c] * (z[i] - meanp[c]) * invp[c] + beta[c] + x[i];
}

extern "C" void kernel_launch(void* const* d_in, const int* /*in_sizes*/, int /*n_in*/,
                              void* d_out, int /*out_size*/, void* d_ws, size_t /*ws_size*/,
                              hipStream_t stream)
{
  const float* x       = (const float*)d_in[0];
  const float* g_w     = (const float*)d_in[1];
  const float* g_b     = (const float*)d_in[2];
  const float* theta_w = (const float*)d_in[3];
  const float* theta_b = (const float*)d_in[4];
  const float* phi_w   = (const float*)d_in[5];
  const float* phi_b   = (const float*)d_in[6];
  const float* w_w     = (const float*)d_in[7];
  const float* w_b     = (const float*)d_in[8];
  const float* bn_g    = (const float*)d_in[9];
  const float* bn_b    = (const float*)d_in[10];
  float* out = (float*)d_out;

  // Workspace layout (aliased regions; lifetimes are stream-ordered):
  //   [0, 64MB): phase1 xT(16MB)+proj(25MB) -> phase2 f(64MB) -> phase4 z(33.5MB)
  char* ws = (char*)d_ws;
  bf16_t* xT     = (bf16_t*)(ws + 0);
  bf16_t* proj   = (bf16_t*)(ws + 16777216);
  bf16_t* fbuf   = (bf16_t*)(ws + 0);
  float*  zbuf   = (float*) (ws + 0);
  bf16_t* thetaT = (bf16_t*)(ws + 67108864);   // dead after f -> reuse for y
  bf16_t* ybuf   = (bf16_t*)(ws + 67108864);
  bf16_t* g_p    = (bf16_t*)(ws + 75497472);
  bf16_t* phiT   = (bf16_t*)(ws + 77594624);
  bf16_t* Wproj  = (bf16_t*)(ws + 79691776);
  bf16_t* Ww     = (bf16_t*)(ws + 79888384);
  float*  bproj  = (float*) (ws + 79953920);
  float*  meanp  = (float*) (ws + 79955968);
  float*  invp   = (float*) (ws + 79956992);

  k_prep_weights<<<514, 256, 0, stream>>>(g_w, g_b, theta_w, theta_b,
                                          phi_w, phi_b, w_w, Wproj, bproj, Ww);
  k_transpose_x<<<dim3(128, 8, 8), 256, 0, stream>>>(x, xT);

  // proj[b][384][4096] = Wproj[384][256] @ x_b[256][4096] + bproj
  wmma_gemm_bf16<false, true><<<dim3(32, 3, 8), 256, 0, stream>>>(
      Wproj, 0, xT, (long)HWN * CCH, proj, (long)384 * HWN, bproj,
      384, HWN, CCH, 1.0f);

  k_theta_T<<<dim3(128, 4, 8), 256, 0, stream>>>(proj, thetaT);
  k_pool_g<<<4096, 256, 0, stream>>>(proj, g_p);
  k_pool_phiT<<<4096, 256, 0, stream>>>(proj, phiT);

  // f[b][4096][1024] = (theta @ phi) / Ns
  wmma_gemm_bf16<false, false><<<dim3(8, 32, 8), 256, 0, stream>>>(
      thetaT, (long)HWN * CIH, phiT, (long)NSS * CIH, fbuf, (long)HWN * NSS,
      nullptr, HWN, NSS, CIH, 1.0f / (float)NSS);

  // y[b][4096][128] = f @ g
  wmma_gemm_bf16<false, false><<<dim3(1, 32, 8), 256, 0, stream>>>(
      fbuf, (long)HWN * NSS, g_p, (long)CIH * NSS, ybuf, (long)HWN * CIH,
      nullptr, HWN, CIH, NSS, 1.0f);

  // z[b][256][4096] = Ww[256][128] @ y_b^T + w_b  (fp32 for BN stats)
  wmma_gemm_bf16<true, true><<<dim3(32, 2, 8), 256, 0, stream>>>(
      Ww, 0, ybuf, (long)HWN * CIH, zbuf, (long)CCH * HWN, w_b,
      CCH, HWN, CIH, 1.0f);

  k_bn_stats<<<256, 256, 0, stream>>>(zbuf, meanp, invp);
  k_bn_apply<<<32768, 256, 0, stream>>>(zbuf, x, bn_g, bn_b, meanp, invp, out);
}
